// Train_34789235098137
// MI455X (gfx1250) — matrix operand order repaired and checked by reading, not compiled, not run on hardware
//
#include <hip/hip_runtime.h>
#include <math.h>

// ---------------------------------------------------------------------------
// Types for CDNA5 WMMA (wave32): v_wmma_f32_16x16x32_bf16
// ---------------------------------------------------------------------------
typedef __attribute__((ext_vector_type(16))) __bf16 v16bf;
typedef __attribute__((ext_vector_type(8)))  __bf16 v8bf;
typedef __attribute__((ext_vector_type(8)))  float  v8f;

union bfrag {
    v16bf v;
    v8bf  h[2];
};

__device__ __forceinline__ int imin(int a, int b) { return a < b ? a : b; }

// Model dims
#define BB   64
#define TT   50
#define LL   21
#define DD   15
#define HH   1100
#define HHP  1120      // 1100 padded to multiple of 32
#define HID  650
#define HIDP 672       // 650 padded to multiple of 32
#define G4   2600      // 4*HID
#define VV   10000
#define NTOK 3200      // B*T
#define NALL 13200     // NTOK + V

// ---------------------------------------------------------------------------
__global__ void zero_u32_kernel(unsigned int* p, long long n) {
    long long i = (long long)blockIdx.x * 256 + threadIdx.x;
    if (i < n) p[i] = 0u;
}

// ---------------------------------------------------------------------------
// f32 -> bf16 row-major with zero padding (activations / A matrices).
// ---------------------------------------------------------------------------
__global__ void f32_to_bf16_pad_kernel(const float* __restrict__ src, int sld,
                                       int vr, int vc,
                                       __bf16* __restrict__ dst, int R, int C) {
    long long i = (long long)blockIdx.x * 256 + threadIdx.x;
    long long n = (long long)R * C;
    if (i >= n) return;
    int r = (int)(i / C), c = (int)(i % C);
    float v = (r < vr && c < vc) ? src[(size_t)r * sld + c] : 0.0f;
    dst[i] = (__bf16)v;
}

// ---------------------------------------------------------------------------
// f32 [K, N] -> bf16 K-major transpose with K zero-padding (B matrices).
// dst [Nv rows][Kp cols]: dst[n*Kp + k] = src[k*sld + n]  (k < Kvalid else 0)
// ---------------------------------------------------------------------------
__global__ void f32_to_bf16_padT_kernel(const float* __restrict__ src, int sld,
                                        int Kvalid, int Nv,
                                        __bf16* __restrict__ dst, int Kp) {
    long long i = (long long)blockIdx.x * 256 + threadIdx.x;
    long long n = (long long)Nv * Kp;
    if (i >= n) return;
    int nr = (int)(i / Kp), k = (int)(i % Kp);
    float v = (k < Kvalid) ? src[(size_t)k * sld + nr] : 0.0f;
    dst[i] = (__bf16)v;
}

// ---------------------------------------------------------------------------
// char embedding + multi-width conv + tanh + max-over-time (VALU; ~12% of
// total FLOPs). One block per row.
// ---------------------------------------------------------------------------
__global__ __launch_bounds__(128) void char_cnn_kernel(
    const int* __restrict__ idx, const float* __restrict__ char_emb,
    const float* __restrict__ f1, const float* __restrict__ b1,
    const float* __restrict__ f2, const float* __restrict__ b2,
    const float* __restrict__ f3, const float* __restrict__ b3,
    const float* __restrict__ f4, const float* __restrict__ b4,
    const float* __restrict__ f5, const float* __restrict__ b5,
    const float* __restrict__ f6, const float* __restrict__ b6,
    const float* __restrict__ f7, const float* __restrict__ b7,
    float* __restrict__ Hout, int rowOffset, int Nrows) {
    int row = blockIdx.x;
    if (row >= Nrows) return;
    __shared__ float emb[LL][16];
    int tid = threadIdx.x;
    for (int i = tid; i < LL * DD; i += 128) {
        int l = i / DD, d = i % DD;
        int ch = idx[(size_t)row * LL + l];
        emb[l][d] = char_emb[(size_t)ch * DD + d];
    }
    __syncthreads();
    const float* filts[7]  = {f1, f2, f3, f4, f5, f6, f7};
    const float* biases[7] = {b1, b2, b3, b4, b5, b6, b7};
    const int off[8] = {0, 50, 150, 300, 500, 700, 900, 1100};
    const int nf[7]  = {50, 100, 150, 200, 200, 200, 200};
    for (int f = tid; f < HH; f += 128) {
        int w = 0;
        while (f >= off[w + 1]) ++w;
        int fi = f - off[w];
        int F = nf[w];
        int width = w + 1;
        const float* filt = filts[w];
        float bias = biases[w][fi];
        float mx = -3.4e38f;
        for (int p = 0; p <= LL - width; ++p) {
            float acc = bias;
            for (int k = 0; k < width; ++k)
                for (int d = 0; d < DD; ++d)
                    acc += emb[p + k][d] * filt[(size_t)(k * DD + d) * F + fi];
            float t = tanhf(acc);
            mx = fmaxf(mx, t);
        }
        Hout[(size_t)(rowOffset + row) * HH + f] = mx;
    }
}

// ---------------------------------------------------------------------------
// bf16 GEMM, B pre-transposed K-major: C[M,N] = A[M,K] @ Bt[N,K]^T (+bias).
// K % 32 == 0. Block tile 128x64, 256 threads = 8 waves; wave w owns rows
// [w*16, w*16+16) x 64 cols = 4 WMMA accumulators.
// Branch-free inner loop: out-of-range rows/cols are CLAMPED, not zeroed —
// they only corrupt outputs that the guarded epilogue discards.
// Software-pipelined: A (global->reg->LDS) and B (global->reg) fetched one
// k-chunk ahead so L2 streams overlap the matrix pipe.
// mode 0: C = acc + bias
// mode 1: highway epilogue: g=sigmoid(gsrc); hio = g*relu(acc+bias)+(1-g)*hio
// mode 2: cb(bf16) = acc + bias   (f32 C not written)
// ---------------------------------------------------------------------------
__global__ __launch_bounds__(256) void gemm_bf16_kernel(
    const __bf16* __restrict__ A, int lda,
    const __bf16* __restrict__ Bt,
    float* __restrict__ C, int ldc,
    const float* __restrict__ bias,
    int M, int N, int K, int mode,
    const float* __restrict__ gsrc,
    float* __restrict__ hio,
    __bf16* __restrict__ cb, int ldcb) {
    __shared__ __align__(16) __bf16 As[128][48];
    int tid  = threadIdx.x;
    int lane = tid & 31;
    int w    = tid >> 5;          // wave id: M sub-row 0..7
    int bm   = blockIdx.y * 128, bn = blockIdx.x * 64;
    int laneN = lane & 15;
    int aoff  = (lane < 16) ? 0 : 16;
    int boff  = (lane < 16) ? 0 : 16;

    v8f accs[4];
#pragma unroll
    for (int h = 0; h < 4; ++h)
#pragma unroll
        for (int i = 0; i < 8; ++i) accs[h][i] = 0.0f;

    // ---- per-thread A staging addresses (rows clamped; 16B chunks) ----
    int i0 = tid, i1 = tid + 256;
    int r0 = i0 >> 2, s0 = i0 & 3;
    int r1 = i1 >> 2, s1 = i1 & 3;
    const __bf16* ap0 = A + (size_t)imin(bm + r0, M - 1) * lda + s0 * 8;
    const __bf16* ap1 = A + (size_t)imin(bm + r1, M - 1) * lda + s1 * 8;
    __bf16* ds0 = &As[r0][s0 * 8];
    __bf16* ds1 = &As[r1][s1 * 8];

    // ---- per-lane B column pointers (cols clamped) ----
    const __bf16* bptr[4];
#pragma unroll
    for (int h = 0; h < 4; ++h) {
        int col = imin(bn + h * 16 + laneN, N - 1);
        bptr[h] = Bt + (size_t)col * K + boff;
    }

    // ---- pipeline prologue: fetch k-chunk 0 ----
    v8bf areg0 = *(const v8bf*)(ap0);
    v8bf areg1 = *(const v8bf*)(ap1);
    v16bf bcur[4];
#pragma unroll
    for (int h = 0; h < 4; ++h) bcur[h] = *(const v16bf*)(bptr[h]);

    for (int k0 = 0; k0 < K; k0 += 32) {
        __syncthreads();
        *(v8bf*)ds0 = areg0;
        *(v8bf*)ds1 = areg1;
        __syncthreads();
        // prefetch next chunk (clamped on last iteration: redundant, not OOB)
        int kn = (k0 + 32 < K) ? (k0 + 32) : k0;
        areg0 = *(const v8bf*)(ap0 + kn);
        areg1 = *(const v8bf*)(ap1 + kn);
        v16bf bnxt[4];
#pragma unroll
        for (int h = 0; h < 4; ++h) bnxt[h] = *(const v16bf*)(bptr[h] + kn);
        // A fragment: two aligned 16B LDS loads
        bfrag a;
        const __bf16* pa = &As[w * 16 + laneN][0];
        a.h[0] = *(const v8bf*)(pa + aoff);
        a.h[1] = *(const v8bf*)(pa + aoff + 8);
#pragma unroll
        for (int h = 0; h < 4; ++h) {
            accs[h] = __builtin_amdgcn_wmma_f32_16x16x32_bf16(
                false, a.v, false, bcur[h], (short)0, accs[h], false, false);
        }
#pragma unroll
        for (int h = 0; h < 4; ++h) bcur[h] = bnxt[h];
    }
    // epilogue; C/D layout: VGPR r -> M = r + (lane<16?0:8), N = lane&15
    int mrow = (lane < 16) ? 0 : 8;
#pragma unroll
    for (int h = 0; h < 4; ++h) {
#pragma unroll
        for (int r = 0; r < 8; ++r) {
            int gm = bm + w * 16 + mrow + r;
            int gn = bn + h * 16 + laneN;
            if (gm >= M || gn >= N) continue;
            float v = accs[h][r];
            if (mode == 0) {
                float bv = bias ? bias[gn] : 0.0f;
                C[(size_t)gm * ldc + gn] = v + bv;
            } else if (mode == 1) {
                size_t ix = (size_t)gm * ldc + gn;
                float g = 1.0f / (1.0f + expf(-gsrc[ix]));
                float a2 = v + bias[gn];
                a2 = a2 > 0.0f ? a2 : 0.0f;
                float hv = hio[ix];
                hio[ix] = g * a2 + (1.0f - g) * hv;
            } else {
                float bv = bias ? bias[gn] : 0.0f;
                cb[(size_t)gm * ldcb + gn] = (__bf16)(v + bv);
            }
        }
    }
}

// ---------------------------------------------------------------------------
// LSTM gate fusion for one time step (TF cell, gate order i,j,f,o, fbias=1).
// ---------------------------------------------------------------------------
__global__ __launch_bounds__(256) void lstm_gates_kernel(
    const float* __restrict__ Zx, const float* __restrict__ Zh,
    float* __restrict__ c, float* __restrict__ Hs,
    __bf16* __restrict__ Hsb, __bf16* __restrict__ hprevb, int t) {
    int i = blockIdx.x * 256 + threadIdx.x;
    if (i >= BB * HID) return;
    int b = i / HID, j = i % HID;
    int n = b * TT + t;
    const float* zx = Zx + (size_t)n * G4;
    const float* zh = Zh + (size_t)b * G4;
    float zi = zx[j] + zh[j];
    float zj = zx[HID + j] + zh[HID + j];
    float zf = zx[2 * HID + j] + zh[2 * HID + j];
    float zo = zx[3 * HID + j] + zh[3 * HID + j];
    float ig = 1.0f / (1.0f + expf(-zi));
    float fg = 1.0f / (1.0f + expf(-(zf + 1.0f)));
    float og = 1.0f / (1.0f + expf(-zo));
    float cn = fg * c[i] + ig * tanhf(zj);
    float hv = og * tanhf(cn);
    c[i] = cn;
    Hs[(size_t)n * HID + j] = hv;
    Hsb[(size_t)n * HIDP + j] = (__bf16)hv;
    hprevb[(size_t)b * HIDP + j] = (__bf16)hv;
}

// ---------------------------------------------------------------------------
// Fused logits + log-softmax NLL. Block = 16 rows of Hs1b [3200,672]bf16
// times Wout^T with Wout [10000,672]bf16 K-major (aligned 32B vector loads).
// Online logsumexp over V, 128 cols per chunk (8 waves x 16 cols).
// OOB vocab columns load clamped garbage; masked to -1e30 before logsumexp.
// ---------------------------------------------------------------------------
__global__ __launch_bounds__(256) void logits_nll_kernel(
    const __bf16* __restrict__ Ab, const __bf16* __restrict__ Wb,
    const float* __restrict__ obias, const int* __restrict__ y,
    float* __restrict__ nll) {
    __shared__ __align__(16) __bf16 Asl[16][HIDP + 8];
    __shared__ float Lc[16][136];
    int tid  = threadIdx.x;
    int lane = tid & 31;
    int w    = tid >> 5;
    int row0 = blockIdx.x * 16;

    // stage A strip as 16B chunks: 16*672/8 = 1344 chunks
    for (int i = tid; i < 16 * (HIDP / 8); i += 256) {
        int r = i / (HIDP / 8), seg = i % (HIDP / 8);
        v8bf vv = *(const v8bf*)(Ab + (size_t)(row0 + r) * HIDP + seg * 8);
        *(v8bf*)(&Asl[r][seg * 8]) = vv;
    }
    __syncthreads();

    float mr = -3.0e38f, sr = 0.0f, tg = 0.0f;
    int ytgt = (tid < 16) ? y[row0 + tid] : -1;

    int am   = lane & 15;
    int aoff = (lane < 16) ? 0 : 16;
    int boff = (lane < 16) ? 0 : 16;

    for (int v0 = 0; v0 < VV; v0 += 128) {
        int colbase = v0 + w * 16;
        int col = colbase + (lane & 15);
        v8f acc;
#pragma unroll
        for (int i = 0; i < 8; ++i) acc[i] = 0.0f;
        const __bf16* bp = Wb + (size_t)imin(col, VV - 1) * HIDP + boff;
#pragma unroll 3
        for (int kk = 0; kk < HIDP; kk += 32) {
            bfrag a;
            const __bf16* pa = &Asl[am][kk];
            a.h[0] = *(const v8bf*)(pa + aoff);
            a.h[1] = *(const v8bf*)(pa + aoff + 8);
            v16bf b = *(const v16bf*)(bp + kk);
            acc = __builtin_amdgcn_wmma_f32_16x16x32_bf16(
                false, a.v, false, b, (short)0, acc, false, false);
        }
        int mrow = (lane < 16) ? 0 : 8;
        int n    = lane & 15;
        int gcol = colbase + n;
#pragma unroll
        for (int r = 0; r < 8; ++r) {
            float val = (gcol < VV) ? (acc[r] + obias[gcol]) : -1.0e30f;
            Lc[mrow + r][w * 16 + n] = val;
        }
        __syncthreads();
        if (tid < 16) {
            for (int cc = 0; cc < 128; ++cc) {
                float val = Lc[tid][cc];
                if (v0 + cc == ytgt) tg = val;
                if (val > mr) {
                    sr = sr * expf(mr - val) + 1.0f;
                    mr = val;
                } else {
                    sr += expf(val - mr);
                }
            }
        }
        __syncthreads();
    }
    if (tid < 16) nll[row0 + tid] = (mr + logf(sr)) - tg;
}

// ---------------------------------------------------------------------------
__global__ __launch_bounds__(256) void nll_reduce_kernel(
    const float* __restrict__ nll, float* __restrict__ out) {
    __shared__ float sm[256];
    float s = 0.0f;
    for (int i = threadIdx.x; i < NTOK; i += 256) s += nll[i];
    sm[threadIdx.x] = s;
    __syncthreads();
    for (int st = 128; st > 0; st >>= 1) {
        if (threadIdx.x < st) sm[threadIdx.x] += sm[threadIdx.x + st];
        __syncthreads();
    }
    if (threadIdx.x == 0) out[0] = sm[0] / (float)BB;
}

// ---------------------------------------------------------------------------
static void launch_gemm(hipStream_t s, const __bf16* A, int lda,
                        const __bf16* Bt, float* C, int ldc,
                        const float* bias, int M, int N, int K, int mode,
                        const float* gsrc, float* hio, __bf16* cb, int ldcb) {
    dim3 g((N + 63) / 64, (M + 127) / 128);
    gemm_bf16_kernel<<<g, dim3(256), 0, s>>>(A, lda, Bt, C, ldc, bias,
                                             M, N, K, mode, gsrc, hio, cb, ldcb);
}

extern "C" void kernel_launch(void* const* d_in, const int* in_sizes, int n_in,
                              void* d_out, int out_size, void* d_ws, size_t ws_size,
                              hipStream_t stream) {
    const int*   x          = (const int*)d_in[0];
    const int*   y          = (const int*)d_in[1];
    const int*   word_chars = (const int*)d_in[2];
    const float* char_emb   = (const float*)d_in[3];
    const float* filt[7];
    const float* cbias[7];
    for (int i = 0; i < 7; ++i) {
        filt[i]  = (const float*)d_in[4 + 2 * i];
        cbias[i] = (const float*)d_in[5 + 2 * i];
    }
    const float* hw1_tw = (const float*)d_in[18];
    const float* hw1_tb = (const float*)d_in[19];
    const float* hw1_hw = (const float*)d_in[20];
    const float* hw1_hb = (const float*)d_in[21];
    const float* hw2_tw = (const float*)d_in[22];
    const float* hw2_tb = (const float*)d_in[23];
    const float* hw2_hw = (const float*)d_in[24];
    const float* hw2_hb = (const float*)d_in[25];
    const float* lstm_k0 = (const float*)d_in[26];
    const float* lstm_b0 = (const float*)d_in[27];
    const float* lstm_k1 = (const float*)d_in[28];
    const float* lstm_b1 = (const float*)d_in[29];
    const float* proj_w  = (const float*)d_in[30];
    const float* out_bias = (const float*)d_in[31];
    (void)in_sizes; (void)n_in; (void)out_size; (void)ws_size;

    // ------------------- workspace layout -------------------
    char* base = (char*)d_ws;
    size_t off = 0;
    auto alloc = [&](size_t bytes) -> void* {
        void* p = base + off;
        off += (bytes + 255) & ~(size_t)255;
        return p;
    };
    float*  Hf   = (float*)alloc((size_t)NALL * HH * 4);      // f32 hidden
    __bf16* Hb   = (__bf16*)alloc((size_t)NALL * HHP * 2);    // bf16 K-padded
    float*  Tbuf = (float*)alloc((size_t)NALL * HH * 4);      // T-gate / Zx
    float*  Zx   = Tbuf;
    float*  Zh   = (float*)alloc((size_t)BB * G4 * 4);
    __bf16* Wtw1 = (__bf16*)alloc((size_t)HH * HHP * 2);      // K-major [N][Kp]
    __bf16* Whw1 = (__bf16*)alloc((size_t)HH * HHP * 2);
    __bf16* Wtw2 = (__bf16*)alloc((size_t)HH * HHP * 2);
    __bf16* Whw2 = (__bf16*)alloc((size_t)HH * HHP * 2);
    __bf16* Wx0  = (__bf16*)alloc((size_t)G4 * HHP * 2);
    __bf16* Wh0  = (__bf16*)alloc((size_t)G4 * HIDP * 2);
    __bf16* Wx1  = (__bf16*)alloc((size_t)G4 * HIDP * 2);
    __bf16* Wh1  = (__bf16*)alloc((size_t)G4 * HIDP * 2);
    __bf16* Pj   = (__bf16*)alloc((size_t)HID * HHP * 2);
    float*  Hs0  = (float*)alloc((size_t)NTOK * HID * 4);
    __bf16* Hs0b = (__bf16*)alloc((size_t)NTOK * HIDP * 2);
    float*  Hs1  = (float*)alloc((size_t)NTOK * HID * 4);
    __bf16* Hs1b = (__bf16*)alloc((size_t)NTOK * HIDP * 2);
    __bf16* Woutb = (__bf16*)alloc((size_t)VV * HIDP * 2);
    float*  cbuf = (float*)alloc((size_t)BB * HID * 4);
    __bf16* hprevb = (__bf16*)alloc((size_t)BB * HIDP * 2);
    float*  nllb = (float*)alloc((size_t)NTOK * 4);

    auto z32 = [&](void* p, size_t n32) {
        zero_u32_kernel<<<(unsigned)((n32 + 255) / 256), 256, 0, stream>>>(
            (unsigned int*)p, (long long)n32);
    };
    auto cvt = [&](const float* src, int sld, int vr, int vc, __bf16* dst,
                   int R, int Cc) {
        size_t n = (size_t)R * Cc;
        f32_to_bf16_pad_kernel<<<(unsigned)((n + 255) / 256), 256, 0, stream>>>(
            src, sld, vr, vc, dst, R, Cc);
    };
    auto cvtT = [&](const float* src, int sld, int Kvalid, int Nv, __bf16* dst,
                    int Kp) {
        size_t n = (size_t)Nv * Kp;
        f32_to_bf16_padT_kernel<<<(unsigned)((n + 255) / 256), 256, 0, stream>>>(
            src, sld, Kvalid, Nv, dst, Kp);
    };

    // zero pad regions that are only partially written later
    z32(Hs0b, (size_t)NTOK * HIDP / 2);
    z32(Hs1b, (size_t)NTOK * HIDP / 2);
    z32(Woutb, (size_t)VV * HIDP / 2);

    // ---- phase 1: char CNN for tokens (rows 0..3199) and vocab (3200..13199)
    char_cnn_kernel<<<NTOK, 128, 0, stream>>>(
        x, char_emb, filt[0], cbias[0], filt[1], cbias[1], filt[2], cbias[2],
        filt[3], cbias[3], filt[4], cbias[4], filt[5], cbias[5], filt[6],
        cbias[6], Hf, 0, NTOK);
    char_cnn_kernel<<<VV, 128, 0, stream>>>(
        word_chars, char_emb, filt[0], cbias[0], filt[1], cbias[1], filt[2],
        cbias[2], filt[3], cbias[3], filt[4], cbias[4], filt[5], cbias[5],
        filt[6], cbias[6], Hf, NTOK, VV);
    cvt(Hf, HH, NALL, HH, Hb, NALL, HHP);

    // ---- weight staging: bf16, transposed to K-major, K padded
    cvtT(hw1_tw, HH, HH, HH, Wtw1, HHP);
    cvtT(hw1_hw, HH, HH, HH, Whw1, HHP);
    cvtT(hw2_tw, HH, HH, HH, Wtw2, HHP);
    cvtT(hw2_hw, HH, HH, HH, Whw2, HHP);
    cvtT(lstm_k0, G4, HH, G4, Wx0, HHP);
    cvtT(lstm_k0 + (size_t)HH * G4, G4, HID, G4, Wh0, HIDP);
    cvtT(lstm_k1, G4, HID, G4, Wx1, HIDP);
    cvtT(lstm_k1 + (size_t)HID * G4, G4, HID, G4, Wh1, HIDP);
    cvtT(proj_w, HID, HH, HID, Pj, HHP);

    // ---- phase 2: two highway layers (WMMA GEMMs + fused gating)
    launch_gemm(stream, Hb, HHP, Wtw1, Tbuf, HH, hw1_tb, NALL, HH, HHP, 0,
                nullptr, nullptr, nullptr, 0);
    launch_gemm(stream, Hb, HHP, Whw1, nullptr, HH, hw1_hb, NALL, HH, HHP, 1,
                Tbuf, Hf, nullptr, 0);
    cvt(Hf, HH, NALL, HH, Hb, NALL, HHP);
    launch_gemm(stream, Hb, HHP, Wtw2, Tbuf, HH, hw2_tb, NALL, HH, HHP, 0,
                nullptr, nullptr, nullptr, 0);
    launch_gemm(stream, Hb, HHP, Whw2, nullptr, HH, hw2_hb, NALL, HH, HHP, 1,
                Tbuf, Hf, nullptr, 0);
    cvt(Hf, HH, NALL, HH, Hb, NALL, HHP);

    // ---- phase 3: LSTM layer 0 (input proj hoisted out of time loop)
    launch_gemm(stream, Hb, HHP, Wx0, Zx, G4, lstm_b0, NTOK, G4, HHP, 0,
                nullptr, nullptr, nullptr, 0);
    z32(cbuf, (size_t)BB * HID);
    z32(hprevb, (size_t)BB * HIDP / 2);
    for (int t = 0; t < TT; ++t) {
        launch_gemm(stream, hprevb, HIDP, Wh0, Zh, G4, nullptr, BB, G4, HIDP,
                    0, nullptr, nullptr, nullptr, 0);
        lstm_gates_kernel<<<(BB * HID + 255) / 256, 256, 0, stream>>>(
            Zx, Zh, cbuf, Hs0, Hs0b, hprevb, t);
    }
    // ---- LSTM layer 1
    launch_gemm(stream, Hs0b, HIDP, Wx1, Zx, G4, lstm_b1, NTOK, G4, HIDP, 0,
                nullptr, nullptr, nullptr, 0);
    z32(cbuf, (size_t)BB * HID);
    z32(hprevb, (size_t)BB * HIDP / 2);
    for (int t = 0; t < TT; ++t) {
        launch_gemm(stream, hprevb, HIDP, Wh1, Zh, G4, nullptr, BB, G4, HIDP,
                    0, nullptr, nullptr, nullptr, 0);
        lstm_gates_kernel<<<(BB * HID + 255) / 256, 256, 0, stream>>>(
            Zx, Zh, cbuf, Hs1, Hs1b, hprevb, t);
    }

    // ---- phase 4: tied softmax weights = H2[vocab rows] @ proj_w  (bf16 out)
    launch_gemm(stream, Hb + (size_t)NTOK * HHP, HHP, Pj, nullptr, HID,
                nullptr, VV, HID, HHP, 2, nullptr, nullptr, Woutb, HIDP);

    // ---- phase 5: fused logits + log-softmax NLL, then reduce
    logits_nll_kernel<<<NTOK / 16, 256, 0, stream>>>(Hs1b, Woutb, out_bias, y,
                                                     nllb);
    nll_reduce_kernel<<<1, 256, 0, stream>>>(nllb, (float*)d_out);
}
